// ALiBiAttention_56573309223554
// MI455X (gfx1250) — matrix operand order repaired and checked by reading, not compiled, not run on hardware
//
#include <hip/hip_runtime.h>
#include <math.h>

typedef __attribute__((ext_vector_type(16))) _Float16 v16h;
typedef __attribute__((ext_vector_type(8)))  float    v8f;

#define S_LEN   2048
#define DHEAD   64
#define KT      32              // k-positions per mainloop iteration
#define NIT     (S_LEN / KT)    // 64 iterations
#define KSTRIDE 72              // halves per ktile row (16B aligned, conflict-free b128 reads)
#define VSTRIDE 40              // halves per vtile(transposed) row (16B aligned)
#define LOG2E   1.4426950408889634f

union Frag16 { v16h v; uint4 u[2]; };

static __device__ __forceinline__ unsigned pkh2(float a, float b) {
    union { _Float16 h[2]; unsigned u; } x;
    x.h[0] = (_Float16)a; x.h[1] = (_Float16)b;
    return x.u;
}

__global__ __launch_bounds__(256)
void alibi_attn_wmma(const float* __restrict__ Q,
                     const float* __restrict__ K,
                     const float* __restrict__ V,
                     float* __restrict__ O)
{
    // double-buffered staging tiles (f16)
    __shared__ __align__(16) _Float16 ktile[2][KT * KSTRIDE];     // [kpos][d]
    __shared__ __align__(16) _Float16 vtile[2][DHEAD * VSTRIDE];  // [d][kpos] (transposed)

    const int tid  = threadIdx.x;
    const int lane = tid & 31;
    const int wave = tid >> 5;
    const int hi   = lane >> 4;      // lane half: 0 or 1
    const int ln   = lane & 15;

    const int bh     = blockIdx.x >> 4;   // b*H + h  (0..31)
    const int head   = bh & 15;
    const int qchunk = blockIdx.x & 15;
    const int qbase  = qchunk * 128 + wave * 16;

    // ALiBi slope for H=16: 2^(-0.5*(h+1)); pre-scaled by log2(e) (base-2 softmax domain)
    const float slope2 = exp2f(-0.5f * (float)(head + 1)) * LOG2E;

    const float* qp = Q + (size_t)bh * S_LEN * DHEAD;
    const float* kp = K + (size_t)bh * S_LEN * DHEAD;
    const float* vp = V + (size_t)bh * S_LEN * DHEAD;
    float*       op = O + (size_t)bh * S_LEN * DHEAD;

    // ---- Q^T B-fragments (loaded once). B layout: lane holds column q = ln,
    // half h <-> d = 8*hi + h (h<8), 16 + 8*hi + (h-8) (h>=8) (+ 32*frag). Scale = log2(e)/sqrt(64) folded in.
    Frag16 qtB[2];
    {
        const float qscale = 0.125f * LOG2E;
        const float* qr = qp + (size_t)(qbase + ln) * DHEAD + hi * 8;
#pragma unroll
        for (int f = 0; f < 2; ++f)
#pragma unroll
            for (int h = 0; h < 16; ++h)
                qtB[f].v[h] = (_Float16)(qr[f * 32 + h + (h & 8)] * qscale);
    }

    // O^T accumulators: 4 d-groups of 16x16 f32 (lane = q column, rows = d)
    v8f acc[4];
#pragma unroll
    for (int g = 0; g < 4; ++g)
#pragma unroll
        for (int e = 0; e < 8; ++e) acc[g][e] = 0.0f;

    float mrow = -INFINITY;   // running max (base-2 domain) for this lane's q column
    float lrow = 0.0f;        // running softmax denominator

    // cooperative staging decomposition: each thread owns (kpos, 8-wide d chunk)
    const int skp = tid >> 3;         // 0..31
    const int sdc = (tid & 7) * 8;    // 0,8,...,56

    // incremental relative-position term: df = qpos - (kbase + 8*hi); bias = -slope2*|df - j|
    float df = (float)(qbase + ln - 8 * hi);

    // ---- prologue: stage tile 0 into buffer 0
    {
        const float4* kr4 = (const float4*)(kp + (size_t)skp * DHEAD + sdc);
        const float4* vr4 = (const float4*)(vp + (size_t)skp * DHEAD + sdc);
        float4 k0 = kr4[0], k1 = kr4[1], v0 = vr4[0], v1 = vr4[1];
        uint4 ku;
        ku.x = pkh2(k0.x, k0.y); ku.y = pkh2(k0.z, k0.w);
        ku.z = pkh2(k1.x, k1.y); ku.w = pkh2(k1.z, k1.w);
        *(uint4*)&ktile[0][skp * KSTRIDE + sdc] = ku;
        const float vf[8] = {v0.x, v0.y, v0.z, v0.w, v1.x, v1.y, v1.z, v1.w};
#pragma unroll
        for (int j = 0; j < 8; ++j)
            vtile[0][(sdc + j) * VSTRIDE + skp] = (_Float16)vf[j];
    }
    __syncthreads();

    for (int it = 0; it < NIT; ++it) {
        const int buf = it & 1;

        // ---- issue next tile's global loads early (overlap with compute)
        float4 nk0, nk1, nv0, nv1;
        if (it + 1 < NIT) {
            const size_t base = (size_t)((it + 1) * KT + skp) * DHEAD + sdc;
            const float4* kr4 = (const float4*)(kp + base);
            const float4* vr4 = (const float4*)(vp + base);
            nk0 = kr4[0]; nk1 = kr4[1]; nv0 = vr4[0]; nv1 = vr4[1];
        }
        if (it + 2 < NIT) {   // warm L2 for the tile after next -> global_prefetch_b8
            const size_t pb = (size_t)((it + 2) * KT + skp) * DHEAD + sdc;
            __builtin_prefetch(kp + pb, 0, 0);
            __builtin_prefetch(vp + pb, 0, 0);
        }

        // ---- T = K * Q^T  (scores transposed). Two 16(kpos)x16(q) tiles.
        v8f t0, t1;
#pragma unroll
        for (int e = 0; e < 8; ++e) { t0[e] = 0.0f; t1[e] = 0.0f; }
#pragma unroll
        for (int dh = 0; dh < 2; ++dh) {
            Frag16 a0, a1;
            const int r0 = (ln)      * KSTRIDE + dh * 32 + 8 * hi;
            const int r1 = (16 + ln) * KSTRIDE + dh * 32 + 8 * hi;
            a0.u[0] = *(const uint4*)&ktile[buf][r0];
            a0.u[1] = *(const uint4*)&ktile[buf][r0 + 16];
            a1.u[0] = *(const uint4*)&ktile[buf][r1];
            a1.u[1] = *(const uint4*)&ktile[buf][r1 + 16];
            t0 = __builtin_amdgcn_wmma_f32_16x16x32_f16(false, a0.v, false, qtB[dh].v,
                                                        (short)0, t0, false, false);
            t1 = __builtin_amdgcn_wmma_f32_16x16x32_f16(false, a1.v, false, qtB[dh].v,
                                                        (short)0, t1, false, false);
        }

        // ---- ALiBi bias (base-2 domain): T -= slope2 * |df - j|, incremental df
#pragma unroll
        for (int j = 0; j < 8; ++j) {
            t0[j] -= slope2 * fabsf(df - (float)j);
            t1[j] -= slope2 * fabsf(df - (float)(j + 16));
        }
        df -= (float)KT;

        // ---- online softmax over kpos (per q column; lane pair l, l^16)
        float mt = -INFINITY;
#pragma unroll
        for (int j = 0; j < 8; ++j) mt = fmaxf(mt, fmaxf(t0[j], t1[j]));
        mt = fmaxf(mt, __shfl_xor(mt, 16, 32));
        const float mnew  = fmaxf(mrow, mt);
        const float alpha = exp2f(mrow - mnew);
        float ssum = 0.0f;
#pragma unroll
        for (int j = 0; j < 8; ++j) {
            t0[j] = exp2f(t0[j] - mnew);
            t1[j] = exp2f(t1[j] - mnew);
            ssum += t0[j] + t1[j];
        }
        ssum += __shfl_xor(ssum, 16, 32);
        lrow = lrow * alpha + ssum;
        mrow = mnew;
#pragma unroll
        for (int g = 0; g < 4; ++g)
#pragma unroll
            for (int e = 0; e < 8; ++e) acc[g][e] *= alpha;

        // ---- build P^T B-fragment: sources are {self, lane^16} only
        Frag16 pB;
#pragma unroll
        for (int j = 0; j < 8; ++j) {
            const float x0 = __shfl_xor(t0[j], 16, 32);
            const float x1 = __shfl_xor(t1[j], 16, 32);
            pB.v[j]     = (_Float16)(hi ? x1    : t0[j]);   // kpos = j      | 16+j
            pB.v[j + 8] = (_Float16)(hi ? t1[j] : x0);      // kpos = j+8    | 24+j
        }

        // ---- O^T += V^T * P^T  (four 16-wide d-groups, contraction over 32 kpos)
#pragma unroll
        for (int g = 0; g < 4; ++g) {
            Frag16 a;
            const int r = (g * 16 + ln) * VSTRIDE + 16 * hi;
            a.u[0] = *(const uint4*)&vtile[buf][r];
            a.u[1] = *(const uint4*)&vtile[buf][r + 8];
            acc[g] = __builtin_amdgcn_wmma_f32_16x16x32_f16(false, a.v, false, pB.v,
                                                            (short)0, acc[g], false, false);
        }

        // ---- stage next tile into the other buffer, then one barrier
        if (it + 1 < NIT) {
            const int nb = (it + 1) & 1;
            uint4 ku;
            ku.x = pkh2(nk0.x, nk0.y); ku.y = pkh2(nk0.z, nk0.w);
            ku.z = pkh2(nk1.x, nk1.y); ku.w = pkh2(nk1.z, nk1.w);
            *(uint4*)&ktile[nb][skp * KSTRIDE + sdc] = ku;
            const float vf[8] = {nv0.x, nv0.y, nv0.z, nv0.w, nv1.x, nv1.y, nv1.z, nv1.w};
#pragma unroll
            for (int j = 0; j < 8; ++j)
                vtile[nb][(sdc + j) * VSTRIDE + skp] = (_Float16)vf[j];
            __syncthreads();
        }
    }

    // ---- normalize and store: O^T tile element (d = g*16 + 8*hi + j, q = ln)
    const float rinv = 1.0f / lrow;
    float* orow = op + (size_t)(qbase + ln) * DHEAD;
#pragma unroll
    for (int g = 0; g < 4; ++g) {
        float4 lo  = make_float4(acc[g][0] * rinv, acc[g][1] * rinv,
                                 acc[g][2] * rinv, acc[g][3] * rinv);
        float4 hi4 = make_float4(acc[g][4] * rinv, acc[g][5] * rinv,
                                 acc[g][6] * rinv, acc[g][7] * rinv);
        *(float4*)&orow[g * 16 + 8 * hi]     = lo;
        *(float4*)&orow[g * 16 + 8 * hi + 4] = hi4;
    }
}

extern "C" void kernel_launch(void* const* d_in, const int* in_sizes, int n_in,
                              void* d_out, int out_size, void* d_ws, size_t ws_size,
                              hipStream_t stream) {
    (void)in_sizes; (void)n_in; (void)out_size; (void)d_ws; (void)ws_size;
    const float* q = (const float*)d_in[0];
    const float* k = (const float*)d_in[1];
    const float* v = (const float*)d_in[2];
    float* out = (float*)d_out;

    // grid: (B*H) * (S/128) = 32 * 16 = 512 blocks, 256 threads (8 wave32)
    dim3 grid(512), block(256);
    hipLaunchKernelGGL(alibi_attn_wmma, grid, block, 0, stream, q, k, v, out);
}
